// ForgetfulCausalTopKAttention_11020886081928
// MI455X (gfx1250) — matrix operand order repaired and checked by reading, not compiled, not run on hardware
//
#include <hip/hip_runtime.h>

typedef __attribute__((ext_vector_type(16))) _Float16 v16h;
typedef __attribute__((ext_vector_type(8)))  float    v8f;

#define B_      2
#define S_      2048
#define HD_     2048
#define NH_     16
#define DH_     128
#define L_      512
#define WINDOW_ 128
#define SINKS_  16
#define KTOP_   64
#define NEGF    (-3.0e38f)
#define SCALE_  0.08838834764831845f   // 128^-0.5

__device__ __forceinline__ void lds_fence() {
    asm volatile("s_wait_dscnt 0" ::: "memory");
}
__device__ __forceinline__ void async_fence() {
    asm volatile("s_wait_asynccnt 0x0" ::: "memory");
}
__device__ __forceinline__ void async_copy_b128(unsigned lds_byte_addr,
                                                const float* gaddr) {
    // GLOBAL_LOAD_ASYNC_TO_LDS_B128: vdst = LDS byte address, vaddr = 64-bit VA
    asm volatile("global_load_async_to_lds_b128 %0, %1, off"
                 :: "v"(lds_byte_addr), "v"(gaddr) : "memory");
}

// ---------------------------------------------------------------------------
// GEMM: C[M,N] = A[M,K] @ B[K,N], f32 in/out, f16 WMMA.
// Block = 128 threads (4 waves) -> 16x64 output tile (wave w owns N-subtile w).
// Per K-step of 32: A(16x32) + B(32x64) f32 tiles staged to LDS with
// GLOBAL_LOAD_ASYNC_TO_LDS_B128 (ASYNCcnt), fragments built from LDS.
// ---------------------------------------------------------------------------
__global__ __launch_bounds__(128)
void gemm_wmma_f16(const float* __restrict__ A, const float* __restrict__ Bm,
                   float* __restrict__ C, int M, int N, int K) {
    __shared__ float As[16 * 32];   // 2 KB
    __shared__ float Bs[32 * 64];   // 8 KB

    const int tid  = threadIdx.x;
    const int wave = tid >> 5;
    const int lane = tid & 31;
    const int half = lane >> 4;
    const int l16  = lane & 15;
    const int nt = blockIdx.x * 64;
    const int mt = blockIdx.y * 16;

    const unsigned asBase = (unsigned)(size_t)As;
    const unsigned bsBase = (unsigned)(size_t)Bs;

    v8f acc = {};
    for (int k0 = 0; k0 < K; k0 += 32) {
        // ---- stage A tile: 16x32 f32 = 128 x 16B chunks, one per thread ----
        {
            const int c   = tid;            // chunk 0..127
            const int row = c >> 3;         // 8 chunks per 32-float row
            const int col = (c & 7) * 4;
            async_copy_b128(asBase + c * 16,
                            A + (size_t)(mt + row) * K + k0 + col);
        }
        // ---- stage B tile: 32x64 f32 = 512 chunks, four per thread ----
#pragma unroll
        for (int i = 0; i < 4; ++i) {
            const int c   = tid + i * 128;  // chunk 0..511
            const int row = c >> 4;         // 16 chunks per 64-float row
            const int col = (c & 15) * 4;
            async_copy_b128(bsBase + c * 16,
                            Bm + (size_t)(k0 + row) * N + nt + col);
        }
        // prefetch next A tile row for this thread
        if (k0 + 32 < K)
            __builtin_prefetch(A + (size_t)(mt + (tid >> 3)) * K + k0 + 32, 0, 1);

        async_fence();
        __syncthreads();

        // ---- fragments from LDS ----
        v16h af, bf;
#pragma unroll
        for (int i = 0; i < 8; ++i)
            af[i]     = (_Float16)As[l16 * 32 + half * 8 + i];
#pragma unroll
        for (int i = 0; i < 8; ++i)
            af[8 + i] = (_Float16)As[l16 * 32 + half * 8 + 16 + i];
#pragma unroll
        for (int i = 0; i < 16; ++i)
            bf[i] = (_Float16)Bs[(half * 8 + (i & 7) + (i >> 3) * 16) * 64 + wave * 16 + l16];

        acc = __builtin_amdgcn_wmma_f32_16x16x32_f16(false, af, false, bf,
                                                     (short)0, acc, false, false);
        __syncthreads();
    }
    // C/D layout: VGPR i -> M = mt + half*8 + i, N = nt + wave*16 + l16
    const int col  = nt + wave * 16 + l16;
    const int orow = mt + half * 8;
#pragma unroll
    for (int i = 0; i < 8; ++i)
        C[(size_t)(orow + i) * N + col] = acc[i];
}

// ---------------------------------------------------------------------------
// Fused attention: per (b, h, 16-query tile).
// Phase 1: WMMA Q·K^T -> LDS scores[16][S] (scaled).
// Phase 2: per row: causal NEG, top-64 over [SINKS, qi-WINDOW], forget gate.
// Phase 3: per row: softmax in LDS, P·V with zero-prob skipping, write o.
// 4 waves / block; wave w owns rows 4w..4w+3 after phase 1.
// ---------------------------------------------------------------------------
__global__ __launch_bounds__(128)
void attn_fused(const float* __restrict__ q, const float* __restrict__ k,
                const float* __restrict__ v,
                const unsigned char* __restrict__ forget,
                float* __restrict__ o) {
    extern __shared__ float smem[];
    float* sc = smem;                                       // [16][S_]
    volatile int*   topi = (volatile int*)(smem + 16 * S_); // [16][KTOP_]
    volatile float* topv = (volatile float*)(topi + 16 * KTOP_);

    const int qt = blockIdx.x, h = blockIdx.y, b = blockIdx.z;
    const int qbase = qt * 16;
    const int tid  = threadIdx.x;
    const int wave = tid >> 5;
    const int lane = tid & 31;
    const int half = lane >> 4;
    const int l16  = lane & 15;

    // ---- Phase 1: scores = scale * q @ k^T (16 x S tile) ----
    v16h afr[4];
    {
        const float* qp = q + ((size_t)b * S_ + qbase + l16) * HD_ + h * DH_;
#pragma unroll
        for (int kc = 0; kc < 4; ++kc) {
            const int kb = kc * 32 + half * 8;
#pragma unroll
            for (int i = 0; i < 8; ++i) afr[kc][i]     = (_Float16)qp[kb + i];
#pragma unroll
            for (int i = 0; i < 8; ++i) afr[kc][8 + i] = (_Float16)qp[kb + 16 + i];
        }
    }
    for (int jt = wave; jt < S_ / 16; jt += 4) {
        const int jbase = jt * 16;
        // B[K=d][N=j]: lane's column j = jbase+l16, table-order d from k row
        const float* kp = k + ((size_t)b * S_ + jbase + l16) * HD_ + h * DH_;
        v8f acc = {};
#pragma unroll
        for (int kc = 0; kc < 4; ++kc) {
            v16h bf;
            const int kb = kc * 32 + half * 8;
#pragma unroll
            for (int i = 0; i < 16; ++i) bf[i] = (_Float16)kp[kb + (i & 7) + (i >> 3) * 16];
            acc = __builtin_amdgcn_wmma_f32_16x16x32_f16(false, afr[kc], false, bf,
                                                         (short)0, acc, false, false);
        }
#pragma unroll
        for (int i = 0; i < 8; ++i)
            sc[(size_t)(half * 8 + i) * S_ + jbase + l16] = acc[i] * SCALE_;
    }
    __syncthreads();

    // ---- Phases 2+3: wave owns 4 rows ----
    for (int r = wave * 4; r < wave * 4 + 4; ++r) {
        const int qi = qbase + r;
        volatile float* row = sc + (size_t)r * S_;

        // causal mask: j > qi -> NEG
        for (int j = qi + 1 + lane; j < S_; j += 32) row[j] = NEGF;

        // candidates outside sinks/window: j in [SINKS, qi-WINDOW]
        const int lo = SINKS_, hi = qi - WINDOW_;
        const int cnt = hi - lo + 1;
        if (cnt > 0) {
            const int kk = cnt < KTOP_ ? cnt : KTOP_;
            for (int t = 0; t < kk; ++t) {
                float bv = NEGF; int bi = -1;
                for (int j = lo + lane; j <= hi; j += 32) {
                    const float s = row[j];
                    if (s > bv) { bv = s; bi = j; }
                }
#pragma unroll
                for (int off = 16; off > 0; off >>= 1) {
                    const float ov = __shfl_xor(bv, off, 32);
                    const int   oi = __shfl_xor(bi, off, 32);
                    if (ov > bv || (ov == bv && oi >= 0 && (bi < 0 || oi < bi))) {
                        bv = ov; bi = oi;
                    }
                }
                // all lanes hold (bv,bi): every lane retires the winner itself
                row[bi] = NEGF;
                if (lane == 0) { topi[r * KTOP_ + t] = bi; topv[r * KTOP_ + t] = bv; }
                lds_fence();
            }
            // drop all non-selected candidates
            for (int j = lo + lane; j <= hi; j += 32) row[j] = NEGF;
            lds_fence();
            // re-insert selected unless stochastically forgotten
            const unsigned char* fm =
                forget + (((size_t)b * NH_ + h) * S_ + qi) * S_;
            for (int t = lane; t < kk; t += 32) {
                const int j = topi[r * KTOP_ + t];
                if (!fm[j]) row[j] = topv[r * KTOP_ + t];
            }
            lds_fence();
        }

        // ---- softmax over j in [0, qi] ----
        float m = NEGF;
        for (int j = lane; j <= qi; j += 32) m = fmaxf(m, row[j]);
#pragma unroll
        for (int off = 16; off > 0; off >>= 1) m = fmaxf(m, __shfl_xor(m, off, 32));
        float lsum = 0.0f;
        for (int j = lane; j <= qi; j += 32) {
            const float s = row[j];
            const float p = (s > NEGF * 0.5f) ? __expf(s - m) : 0.0f;
            row[j] = p;
            lsum += p;
        }
#pragma unroll
        for (int off = 16; off > 0; off >>= 1) lsum += __shfl_xor(lsum, off, 32);
        const float inv = 1.0f / lsum;
        lds_fence();

        // ---- P @ V: lane covers d = lane*4 .. lane*4+3; skip p == 0 ----
        float a0 = 0.f, a1 = 0.f, a2 = 0.f, a3 = 0.f;
        const float* vb = v + (size_t)b * S_ * HD_ + h * DH_ + lane * 4;
        for (int j = 0; j <= qi; ++j) {
            const float p = row[j];              // uniform across lanes
            if (p != 0.0f) {
                const float* vp = vb + (size_t)j * HD_;
                a0 += p * vp[0]; a1 += p * vp[1];
                a2 += p * vp[2]; a3 += p * vp[3];
            }
        }
        float* op = o + ((size_t)b * S_ + qi) * HD_ + h * DH_ + lane * 4;
        op[0] = a0 * inv; op[1] = a1 * inv; op[2] = a2 * inv; op[3] = a3 * inv;
    }
}

// ---------------------------------------------------------------------------
extern "C" void kernel_launch(void* const* d_in, const int* in_sizes, int n_in,
                              void* d_out, int out_size, void* d_ws, size_t ws_size,
                              hipStream_t stream) {
    const float* x  = (const float*)d_in[0];
    const float* Wq = (const float*)d_in[1];
    const float* Wc = (const float*)d_in[2];
    const float* Wk = (const float*)d_in[3];
    const float* Wv = (const float*)d_in[4];
    const float* Wo = (const float*)d_in[5];
    const unsigned char* forget = (const unsigned char*)d_in[6];
    float* out = (float*)d_out;

    // workspace carve-up (f32): q | c | k | v | o
    float* q = (float*)d_ws;
    float* c = q + (size_t)B_ * S_ * HD_;
    float* k = c + (size_t)B_ * S_ * L_;
    float* v = k + (size_t)B_ * S_ * HD_;
    float* o = v + (size_t)B_ * S_ * HD_;

    const int M = B_ * S_;      // 4096
    dim3 gblk(128);

    // q = x @ Wq            [M,HD] = [M,HD] @ [HD,HD]
    gemm_wmma_f16<<<dim3(HD_ / 64, M / 16), gblk, 0, stream>>>(x, Wq, q, M, HD_, HD_);
    // c = x @ Wc            [M,L]  = [M,HD] @ [HD,L]
    gemm_wmma_f16<<<dim3(L_  / 64, M / 16), gblk, 0, stream>>>(x, Wc, c, M, L_, HD_);
    // k = c @ Wk            [M,HD] = [M,L] @ [L,HD]
    gemm_wmma_f16<<<dim3(HD_ / 64, M / 16), gblk, 0, stream>>>(c, Wk, k, M, HD_, L_);
    // v = c @ Wv
    gemm_wmma_f16<<<dim3(HD_ / 64, M / 16), gblk, 0, stream>>>(c, Wv, v, M, HD_, L_);

    // fused attention: scores in LDS (16*S f32 = 128 KB) + topk scratch (8 KB)
    const size_t shmem = (size_t)16 * S_ * sizeof(float)
                       + (size_t)16 * KTOP_ * (sizeof(int) + sizeof(float));
    attn_fused<<<dim3(S_ / 16, NH_, B_), dim3(128), shmem, stream>>>(q, k, v, forget, o);

    // out = o @ Wo
    gemm_wmma_f16<<<dim3(HD_ / 64, M / 16), gblk, 0, stream>>>(o, Wo, out, M, HD_, HD_);
}
